// DecoderLSTM_63385127355040
// MI455X (gfx1250) — compile-verified, matrix operand order checked by reading, not run on hardware
//
#include <hip/hip_runtime.h>
#include <hip/hip_bf16.h>
#include <math.h>

// ---------------------------------------------------------------------------
// DecoderLSTM for MI455X (gfx1250): persistent wave32 WMMA kernel.
//   B=256, T=64, S=256, V=512, E=512, H=1024, ENC=1024, SUM=2048
// Grid = 16 WGs, each owns a 16-row batch tile for all 64 timesteps.
// All dense layers run as bf16 v_wmma_f32_16x16x32_bf16 with f32 accumulate.
// Weights bf16 (~40MB -> L2 resident); encoder memory optionally bf16
// (134MB -> fully L2 resident) so per-step attention never touches HBM.
// Activations use hw transcendentals (v_tanh_f32 / v_exp_f32 / v_rcp_f32).
// ---------------------------------------------------------------------------

typedef __bf16 bf16;
typedef __attribute__((ext_vector_type(16))) __bf16 bf16x16;
typedef __attribute__((ext_vector_type(8)))  __bf16 bf16x8;
typedef __attribute__((ext_vector_type(8)))  float  v8f;
typedef __attribute__((ext_vector_type(4)))  float  f32x4;

#define DEVI static __device__ __forceinline__

constexpr int B_ = 256, T_ = 64, S_ = 256, V_ = 512, E_ = 512;
constexpr int H_ = 1024, ENC_ = 1024, SUM_ = 2048;
constexpr int MT   = 16;    // batch rows per WG (WMMA M)
constexpr int XSTR = 2048;  // bf16 activation staging row stride (max K)
constexpr int OSTR = 1024;  // f32 output staging row stride
constexpr int NW   = 32;    // waves per block in decoder (1024 threads)

// LDS: xbuf(16x2048 bf16) h0b/h1b/crossb(16x1024 bf16) obuf(16x1024 f32) sbuf(16x256 f32)
constexpr unsigned SMEM_BYTES =
    MT * XSTR * 2 + 3 * (MT * 1024 * 2) + MT * OSTR * 4 + MT * S_ * 4; // 245760 B

DEVI v8f vzero() { v8f z; for (int i = 0; i < 8; ++i) z[i] = 0.f; return z; }

// Branchless hw-transcendental activations (v_tanh_f32 on gfx1250).
DEVI float fast_tanh(float x) {
#if __has_builtin(__builtin_amdgcn_tanhf)
  return __builtin_amdgcn_tanhf(x);
#else
  // tanh(x) = 1 - 2/(exp2(2x*log2e)+1)
  float e = __builtin_amdgcn_exp2f(x * 2.8853900817779268f);
  return 1.f - 2.f * __builtin_amdgcn_rcpf(e + 1.f);
#endif
}
DEVI float fast_sig(float x) {
  // sigmoid(x) = 1/(1+exp2(-x*log2e))
  return __builtin_amdgcn_rcpf(1.f + __builtin_amdgcn_exp2f(x * -1.4426950408889634f));
}

DEVI v8f wmma_bf(bf16x16 a, bf16x16 b, v8f c) {
  return __builtin_amdgcn_wmma_f32_16x16x32_bf16(false, a, false, b, (short)0, c,
                                                 false, false);
}

// A fragment (16x32 bf16) from LDS, row-major, row stride XSTR.
// Lane L: M = L%16;  K groups {kb..kb+7, kb+16..kb+23}, kb = (L/16)*8.
DEVI bf16x16 ldA(const bf16* x, int k0) {
  int lane = threadIdx.x & 31;
  int m  = lane & 15;
  int kb = (lane >> 4) << 3;
  bf16x8 lo = *(const bf16x8*)(x + m * XSTR + k0 + kb);
  bf16x8 hi = *(const bf16x8*)(x + m * XSTR + k0 + kb + 16);
  return __builtin_shufflevector(lo, hi, 0, 1, 2, 3, 4, 5, 6, 7,
                                 8, 9, 10, 11, 12, 13, 14, 15);
}

// B fragment (32x16 bf16) = tile of W^T where W is (N x K) row-major bf16.
// B[k][n] = W[n][k]: lane L holds N = tn*16 + L%16, contiguous K runs.
DEVI bf16x16 ldB(const bf16* W, int K, int tn, int k0) {
  int lane = threadIdx.x & 31;
  int n  = tn * 16 + (lane & 15);
  int kb = (lane >> 4) << 3;
  const bf16* row = W + (size_t)n * K + k0 + kb;
  bf16x8 lo = *(const bf16x8*)(row);
  bf16x8 hi = *(const bf16x8*)(row + 16);
  return __builtin_shufflevector(lo, hi, 0, 1, 2, 3, 4, 5, 6, 7,
                                 8, 9, 10, 11, 12, 13, 14, 15);
}

// Generic Y(16xN) = act(X(16xK) @ W^T + bias) -> f32 LDS (and optional global).
DEVI void gemm_tiles(const bf16* xlds, int K, const bf16* W, const float* bias,
                     int N, float* olds, float* gout, int act_tanh, int t, int r0) {
  int lane = threadIdx.x & 31, wave = threadIdx.x >> 5;
  int n16 = lane & 15, mb = (lane >> 4) << 3;
  int ntiles = N >> 4;
  for (int tn = wave; tn < ntiles; tn += NW) {
    v8f acc = vzero();
#pragma unroll 4
    for (int k0 = 0; k0 < K; k0 += 32)
      acc = wmma_bf(ldA(xlds, k0), ldB(W, K, tn, k0), acc);
    int n = tn * 16 + n16;
    float bv = bias[n];
    for (int e = 0; e < 8; ++e) {
      float v = acc[e] + bv;
      if (act_tanh) v = fast_tanh(v);
      int m = mb + e;
      olds[m * OSTR + n] = v;
      if (gout) gout[((size_t)(r0 + m) * T_ + t) * V_ + n] = v;
    }
  }
}

// One LSTM layer: gates = X(16xK) @ Wg^T (Wg = [Wih|Whh], 4096 x K), then the
// cell update fully in registers. Wave owns hidden-unit tiles {wave, wave+32},
// so it computes i/f/g/o for its own units. Writes new h (bf16) to hlds and to
// xdst cols [0,1024); optionally new c (bf16) to xdst cols [1024,2048).
DEVI void lstm_layer(const bf16* xlds, int K, const bf16* Wg,
                     const float* bih, const float* bhh,
                     v8f* cf, bf16* hlds, bf16* xdst, bool write_c) {
  int lane = threadIdx.x & 31, wave = threadIdx.x >> 5;
  int n16 = lane & 15, mb = (lane >> 4) << 3;
  v8f hf[2];
  for (int qq = 0; qq < 2; ++qq) {
    int ti = wave + qq * 32;
    v8f ai = vzero(), af = vzero(), ag = vzero(), ao = vzero();
#pragma unroll 2
    for (int k0 = 0; k0 < K; k0 += 32) {
      bf16x16 a = ldA(xlds, k0);
      ai = wmma_bf(a, ldB(Wg, K, ti, k0), ai);
      af = wmma_bf(a, ldB(Wg, K, 64 + ti, k0), af);
      ag = wmma_bf(a, ldB(Wg, K, 128 + ti, k0), ag);
      ao = wmma_bf(a, ldB(Wg, K, 192 + ti, k0), ao);
    }
    int u = ti * 16 + n16;
    float bi = bih[u] + bhh[u];
    float bf2 = bih[1024 + u] + bhh[1024 + u];
    float bg = bih[2048 + u] + bhh[2048 + u];
    float bo = bih[3072 + u] + bhh[3072 + u];
    for (int e = 0; e < 8; ++e) {
      float ii = fast_sig(ai[e] + bi);
      float ff = fast_sig(af[e] + bf2);
      float gg = fast_tanh(ag[e] + bg);
      float oo = fast_sig(ao[e] + bo);
      float cc = ff * cf[qq][e] + ii * gg;
      cf[qq][e] = cc;
      hf[qq][e] = oo * fast_tanh(cc);
    }
  }
  __syncthreads();  // all A reads of xlds done before overwriting xdst
  for (int qq = 0; qq < 2; ++qq) {
    int u = (wave + qq * 32) * 16 + n16;
    for (int e = 0; e < 8; ++e) {
      int m = mb + e;
      bf16 hb = (bf16)hf[qq][e];
      hlds[m * 1024 + u] = hb;
      xdst[m * XSTR + u] = hb;
      if (write_c) xdst[m * XSTR + 1024 + u] = (bf16)cf[qq][e];
    }
  }
}

// --------------------------- weight pack / convert --------------------------
__global__ void pack_bf16(const float* __restrict__ A, const float* __restrict__ Bp,
                          bf16* __restrict__ dst, int N, int Ka, int Kb) {
  int K = Ka + Kb;
  long total = (long)N * K;
  for (long i = (long)blockIdx.x * blockDim.x + threadIdx.x; i < total;
       i += (long)gridDim.x * blockDim.x) {
    int r = (int)(i / K), c = (int)(i % K);
    float v = (c < Ka) ? A[(long)r * Ka + c] : Bp[(long)r * Kb + (c - Ka)];
    dst[i] = (bf16)v;
  }
}

// ------------------------------ decoder init --------------------------------
// hcinit(256x4096) = tanh(summary_bf16(256x2048) @ W_init^T + b_init)
__global__ __launch_bounds__(512) void init_state_kernel(
    const bf16* __restrict__ sumb, const bf16* __restrict__ winit,
    const float* __restrict__ binit, float* __restrict__ hcinit) {
  __shared__ bf16 As[MT * XSTR];  // 64KB
  int tid = threadIdx.x, lane = tid & 31, wave = tid >> 5;
  int r0 = blockIdx.x * MT;
  for (int i = tid; i < MT * SUM_; i += 512) {
    int m = i >> 11, c = i & 2047;
    As[m * XSTR + c] = sumb[(size_t)(r0 + m) * SUM_ + c];
  }
  __syncthreads();
  int n16 = lane & 15, mb = (lane >> 4) << 3;
  for (int tn = wave; tn < 256; tn += 16) {
    v8f acc = vzero();
#pragma unroll 4
    for (int k0 = 0; k0 < SUM_; k0 += 32)
      acc = wmma_bf(ldA(As, k0), ldB(winit, SUM_, tn, k0), acc);
    int n = tn * 16 + n16;
    float bv = binit[n];
    for (int e = 0; e < 8; ++e)
      hcinit[(size_t)(r0 + mb + e) * 4096 + n] = fast_tanh(acc[e] + bv);
  }
}

// --------------------------- persistent decoder -----------------------------
__global__ __launch_bounds__(1024) void decoder_persistent(
    const int* __restrict__ y, const float* __restrict__ enc,
    const bf16* __restrict__ encb,  // bf16 copy of enc (may be null)
    const bf16* __restrict__ embb, const bf16* __restrict__ Wci,
    const float* __restrict__ bci, const bf16* __restrict__ W0,
    const float* __restrict__ bih0, const float* __restrict__ bhh0,
    const bf16* __restrict__ W1, const float* __restrict__ bih1,
    const float* __restrict__ bhh1, const bf16* __restrict__ Wq,
    const float* __restrict__ bq, const bf16* __restrict__ Wco,
    const float* __restrict__ bco, const bf16* __restrict__ Wout,
    const float* __restrict__ bout, const float* __restrict__ hcinit,
    int* __restrict__ tok_out, float* __restrict__ logit_out) {
  extern __shared__ char smem[];
  bf16* xbuf   = (bf16*)smem;            // 16 x 2048
  bf16* h0b    = xbuf + MT * XSTR;       // 16 x 1024
  bf16* h1b    = h0b + MT * 1024;
  bf16* crossb = h1b + MT * 1024;
  float* obuf  = (float*)(crossb + MT * 1024);  // 16 x 1024 f32
  float* sbuf  = obuf + MT * OSTR;              // 16 x 256  f32

  const int tid = threadIdx.x, lane = tid & 31, wave = tid >> 5;
  const int r0 = blockIdx.x * MT;
  const int n16 = lane & 15, mb = (lane >> 4) << 3;

  // cell state kept in C/D-fragment register layout; wave owns tiles wave,wave+32
  v8f c0f[2], c1f[2];
  for (int qq = 0; qq < 2; ++qq) {
    int u = (wave + qq * 32) * 16 + n16;
    for (int e = 0; e < 8; ++e) {
      size_t row = (size_t)(r0 + mb + e) * 4096;
      c0f[qq][e] = hcinit[row + 2048 + u];
      c1f[qq][e] = hcinit[row + 3072 + u];
    }
  }
  for (int i = tid; i < MT * 1024; i += 1024) {
    int m = i >> 10, c = i & 1023;
    size_t row = (size_t)(r0 + m) * 4096;
    h0b[i] = (bf16)hcinit[row + c];
    h1b[i] = (bf16)hcinit[row + 1024 + c];
    crossb[i] = (bf16)0.f;
  }
  __syncthreads();

  for (int t = 0; t < T_; ++t) {
    // P0: xbuf = [emb(tok) | cross]
    for (int i = tid; i < MT * E_; i += 1024) {
      int m = i >> 9, c = i & 511;
      int tok = y[(r0 + m) * T_ + t];
      xbuf[m * XSTR + c] = embb[tok * E_ + c];
    }
    for (int i = tid; i < MT * 1024; i += 1024) {
      int m = i >> 10, c = i & 1023;
      xbuf[m * XSTR + 512 + c] = crossb[i];
    }
    __syncthreads();
    // P1: emb' = tanh(xcat @ W_ci^T + b_ci)
    gemm_tiles(xbuf, 1536, Wci, bci, 512, obuf, nullptr, 1, t, r0);
    __syncthreads();
    // P2: xbuf = [emb' | h0]
    for (int i = tid; i < MT * 512; i += 1024) {
      int m = i >> 9, c = i & 511;
      xbuf[m * XSTR + c] = (bf16)obuf[m * OSTR + c];
    }
    for (int i = tid; i < MT * 1024; i += 1024) {
      int m = i >> 10, c = i & 1023;
      xbuf[m * XSTR + 512 + c] = h0b[i];
    }
    __syncthreads();
    // P3: LSTM layer 0 (writes h_l0 -> h0b and xbuf[:,0:1024))
    lstm_layer(xbuf, 1536, W0, bih0, bhh0, c0f, h0b, xbuf, false);
    __syncthreads();
    // P4: xbuf[:,1024:) = h1(old)
    for (int i = tid; i < MT * 1024; i += 1024) {
      int m = i >> 10, c = i & 1023;
      xbuf[m * XSTR + 1024 + c] = h1b[i];
    }
    __syncthreads();
    // P5: LSTM layer 1 (writes h_l1 -> h1b, xbuf[:,0:1024); c_l1 -> xbuf[:,1024:))
    lstm_layer(xbuf, 2048, W1, bih1, bhh1, c1f, h1b, xbuf, true);
    __syncthreads();
    // P6: q = tanh([h_l1|c_l1] @ Wq^T + bq)  -> obuf f32
    gemm_tiles(xbuf, 2048, Wq, bq, 1024, obuf, nullptr, 1, t, r0);
    __syncthreads();
    // P7: attention scores; enc rows L2-resident when bf16 path is active
    for (int p = tid; p < MT * S_; p += 1024) {
      int r = p >> 8, s = p & 255;
      const float* qp = obuf + r * OSTR;
      float dot = 0.f;
      if (encb) {
        const bf16* ep = encb + ((size_t)(r0 + r) * S_ + s) * ENC_;
        for (int e = 0; e < ENC_; e += 8) {
          bf16x8 ev = *(const bf16x8*)(ep + e);
          for (int i = 0; i < 8; ++i) dot += (float)ev[i] * qp[e + i];
        }
      } else {
        const float* ep = enc + ((size_t)(r0 + r) * S_ + s) * ENC_;
        f32x4 accv = {0.f, 0.f, 0.f, 0.f};
        for (int e = 0; e < ENC_; e += 4) {
          f32x4 ev = *(const f32x4*)(ep + e);
          f32x4 qv = *(const f32x4*)(qp + e);
          accv += ev * qv;
        }
        dot = accv[0] + accv[1] + accv[2] + accv[3];
      }
      sbuf[r * S_ + s] = dot;
    }
    __syncthreads();
    // P8: softmax per row (one wave per row, wave32 shuffle reductions)
    if (wave < MT) {
      float mx = -1e30f;
      for (int j = 0; j < 8; ++j) mx = fmaxf(mx, sbuf[wave * S_ + lane + j * 32]);
      for (int off = 16; off >= 1; off >>= 1) mx = fmaxf(mx, __shfl_xor(mx, off, 32));
      float sum = 0.f;
      for (int j = 0; j < 8; ++j) {
        float ev = __builtin_amdgcn_exp2f((sbuf[wave * S_ + lane + j * 32] - mx) *
                                          1.4426950408889634f);
        sbuf[wave * S_ + lane + j * 32] = ev;
        sum += ev;
      }
      for (int off = 16; off >= 1; off >>= 1) sum += __shfl_xor(sum, off, 32);
      float inv = __builtin_amdgcn_rcpf(sum);
      for (int j = 0; j < 8; ++j) sbuf[wave * S_ + lane + j * 32] *= inv;
    }
    __syncthreads();
    // P9: cross = attn @ enc  (64 threads per row, 16 f32 accumulators each)
    {
      int r = tid >> 6, e0 = (tid & 63) * 16;
      float acc[16];
      for (int i = 0; i < 16; ++i) acc[i] = 0.f;
      if (encb) {
        const bf16* erow = encb + (size_t)(r0 + r) * S_ * ENC_ + e0;
        for (int s = 0; s < S_; ++s) {
          float w = sbuf[r * S_ + s];
          const bf16x8* ep = (const bf16x8*)(erow + (size_t)s * ENC_);
          bf16x8 v0 = ep[0], v1 = ep[1];
          for (int i = 0; i < 8; ++i) {
            acc[i]     += (float)v0[i] * w;
            acc[8 + i] += (float)v1[i] * w;
          }
        }
      } else {
        const float* erow = enc + (size_t)(r0 + r) * S_ * ENC_ + e0;
        for (int s = 0; s < S_; ++s) {
          float w = sbuf[r * S_ + s];
          const f32x4* ep = (const f32x4*)(erow + (size_t)s * ENC_);
          for (int j = 0; j < 4; ++j) {
            f32x4 v = ep[j];
            for (int i = 0; i < 4; ++i) acc[j * 4 + i] += v[i] * w;
          }
        }
      }
      for (int i = 0; i < 16; ++i) crossb[r * 1024 + e0 + i] = (bf16)acc[i];
    }
    __syncthreads();
    // P10: xbuf = [h_l1 | cross]  (h_l1 still in xbuf[:,0:1024))
    for (int i = tid; i < MT * 1024; i += 1024) {
      int m = i >> 10, c = i & 1023;
      xbuf[m * XSTR + 1024 + c] = crossb[i];
    }
    __syncthreads();
    // P11: lin = tanh([h_l1|cross] @ Wco^T + bco)
    gemm_tiles(xbuf, 2048, Wco, bco, 1024, obuf, nullptr, 1, t, r0);
    __syncthreads();
    // P12: xbuf[:,0:1024) = bf16(lin)
    for (int i = tid; i < MT * 1024; i += 1024) {
      int m = i >> 10, c = i & 1023;
      xbuf[m * XSTR + c] = (bf16)obuf[m * OSTR + c];
    }
    __syncthreads();
    // P13: logits = lin @ Wout^T + bout -> obuf + global (B,T,V)
    gemm_tiles(xbuf, 1024, Wout, bout, 512, obuf, logit_out, 0, t, r0);
    __syncthreads();
    // P14: argmax over V=512 (first-index tie-break), one wave per row
    if (wave < MT) {
      float best = -1e30f;
      int bidx = 0;
      for (int j = 0; j < 16; ++j) {
        int s2 = lane + j * 32;
        float v = obuf[wave * OSTR + s2];
        if (v > best) { best = v; bidx = s2; }
      }
      for (int off = 16; off >= 1; off >>= 1) {
        float ob = __shfl_xor(best, off, 32);
        int oi = __shfl_xor(bidx, off, 32);
        if (ob > best || (ob == best && oi < bidx)) { best = ob; bidx = oi; }
      }
      if (lane == 0) tok_out[(r0 + wave) * T_ + t] = bidx;
    }
    __syncthreads();
  }
}

// ------------------------------- host launch --------------------------------
extern "C" void kernel_launch(void* const* d_in, const int* in_sizes, int n_in,
                              void* d_out, int out_size, void* d_ws, size_t ws_size,
                              hipStream_t stream) {
  const int*   y     = (const int*)d_in[0];
  const float* enc   = (const float*)d_in[1];
  const float* esum  = (const float*)d_in[2];
  const float* embt  = (const float*)d_in[3];
  const float* W_init = (const float*)d_in[4];
  const float* b_init = (const float*)d_in[5];
  const float* W_ci  = (const float*)d_in[6];
  const float* b_ci  = (const float*)d_in[7];
  const float* Wih0  = (const float*)d_in[8];
  const float* Whh0  = (const float*)d_in[9];
  const float* bih0  = (const float*)d_in[10];
  const float* bhh0  = (const float*)d_in[11];
  const float* Wih1  = (const float*)d_in[12];
  const float* Whh1  = (const float*)d_in[13];
  const float* bih1  = (const float*)d_in[14];
  const float* bhh1  = (const float*)d_in[15];
  const float* Wq    = (const float*)d_in[16];
  const float* bq    = (const float*)d_in[17];
  const float* Wco   = (const float*)d_in[18];
  const float* bco   = (const float*)d_in[19];
  const float* Wout  = (const float*)d_in[20];
  const float* bout  = (const float*)d_in[21];

  // Workspace carve (bf16 packed weights + f32 init state + optional bf16 enc)
  char* w = (char*)d_ws;
  auto take = [&](size_t bytes) { char* p = w; w += (bytes + 255) & ~size_t(255); return p; };
  bf16* Wci_b  = (bf16*)take((size_t)512 * 1536 * 2);
  bf16* W0_b   = (bf16*)take((size_t)4096 * 1536 * 2);
  bf16* W1_b   = (bf16*)take((size_t)4096 * 2048 * 2);
  bf16* Wq_b   = (bf16*)take((size_t)1024 * 2048 * 2);
  bf16* Wco_b  = (bf16*)take((size_t)1024 * 2048 * 2);
  bf16* Wout_b = (bf16*)take((size_t)512 * 1024 * 2);
  bf16* emb_b  = (bf16*)take((size_t)512 * 512 * 2);
  bf16* sum_b  = (bf16*)take((size_t)256 * 2048 * 2);
  bf16* Winit_b = (bf16*)take((size_t)4096 * 2048 * 2);
  float* hcinit = (float*)take((size_t)256 * 4096 * 4);
  // bf16 encoder memory: 134MB -> fits global L2 (192MB); use only if ws allows
  const size_t encBytes = (size_t)B_ * S_ * ENC_ * 2;
  bf16* enc_b = nullptr;
  if ((size_t)(w - (char*)d_ws) + encBytes + 256 <= ws_size)
    enc_b = (bf16*)take(encBytes);
  (void)in_sizes; (void)n_in; (void)out_size;

  dim3 pg(1024), pb(256);
  hipLaunchKernelGGL(pack_bf16, pg, pb, 0, stream, W_ci, W_ci, Wci_b, 512, 1536, 0);
  hipLaunchKernelGGL(pack_bf16, pg, pb, 0, stream, Wih0, Whh0, W0_b, 4096, 512, 1024);
  hipLaunchKernelGGL(pack_bf16, pg, pb, 0, stream, Wih1, Whh1, W1_b, 4096, 1024, 1024);
  hipLaunchKernelGGL(pack_bf16, pg, pb, 0, stream, Wq, Wq, Wq_b, 1024, 2048, 0);
  hipLaunchKernelGGL(pack_bf16, pg, pb, 0, stream, Wco, Wco, Wco_b, 1024, 2048, 0);
  hipLaunchKernelGGL(pack_bf16, pg, pb, 0, stream, Wout, Wout, Wout_b, 512, 1024, 0);
  hipLaunchKernelGGL(pack_bf16, pg, pb, 0, stream, embt, embt, emb_b, 512, 512, 0);
  hipLaunchKernelGGL(pack_bf16, pg, pb, 0, stream, esum, esum, sum_b, 256, 2048, 0);
  hipLaunchKernelGGL(pack_bf16, pg, pb, 0, stream, W_init, W_init, Winit_b, 4096, 2048, 0);
  if (enc_b)
    hipLaunchKernelGGL(pack_bf16, dim3(4096), pb, 0, stream, enc, enc, enc_b,
                       B_ * S_, ENC_, 0);

  hipLaunchKernelGGL(init_state_kernel, dim3(B_ / MT), dim3(512), 0, stream,
                     sum_b, Winit_b, b_init, hcinit);

  int*   tok_out  = (int*)d_out;                      // (B,T) int32 tokens
  float* logits   = (float*)d_out + (size_t)B_ * T_;  // (B,T,V) f32

  (void)hipFuncSetAttribute((const void*)decoder_persistent,
                            hipFuncAttributeMaxDynamicSharedMemorySize,
                            (int)SMEM_BYTES);
  hipLaunchKernelGGL(decoder_persistent, dim3(B_ / MT), dim3(1024), SMEM_BYTES,
                     stream, y, enc, enc_b, emb_b, Wci_b, b_ci, W0_b, bih0, bhh0,
                     W1_b, bih1, bhh1, Wq_b, bq, Wco_b, bco, Wout_b, bout, hcinit,
                     tok_out, logits);
}
